// RNN_69483980914793
// MI455X (gfx1250) — matrix operand order repaired and checked by reading, compiled, not run on hardware
//
#include <hip/hip_runtime.h>

// ---------------------------------------------------------------------------
// Problem constants (from reference): B=64, T=1024, D_IN=128, D_LAT=512, D_OUT=128
// ---------------------------------------------------------------------------
#define RB    64
#define RT    1024
#define DIN   128
#define DLAT  512
#define DOUT  128

typedef __attribute__((ext_vector_type(16))) __bf16 v16bf;
typedef __attribute__((ext_vector_type(8)))  float  v8f;

// round-to-nearest-even f32 -> bf16 (bit pattern)
__device__ __forceinline__ unsigned short f2bf(float f) {
    unsigned u = __float_as_uint(f);
    u += 0x7FFFu + ((u >> 16) & 1u);
    return (unsigned short)(u >> 16);
}
__device__ __forceinline__ unsigned pack2(float lo, float hi) {
    return ((unsigned)f2bf(hi) << 16) | (unsigned)f2bf(lo);
}

__device__ __forceinline__ v8f wmma_bf16(v16bf a, v16bf b, v8f c) {
    // v_wmma_f32_16x16x32_bf16 : D = A(16x32) * B(32x16) + C(16x16)
    return __builtin_amdgcn_wmma_f32_16x16x32_bf16(
        /*neg_a=*/false, a, /*neg_b=*/false, b,
        /*c_mod=*/(short)0, c, /*reuse_a=*/false, /*reuse_b=*/false);
}

// ---------------------------------------------------------------------------
// A-fragment (16x32 bf16). ISA layout: lane m = lane&15, half = lane>>4;
//   elements 0..7  <-> K = half*8 + 0..7
//   elements 8..15 <-> K = 16 + half*8 + 0..7
// ---------------------------------------------------------------------------
__device__ __forceinline__ v16bf load_afrag_bf16(const unsigned short* base, int ld,
                                                 int lane) {
    const int m = lane & 15, half = lane >> 4;
    const unsigned short* r = base + (long)m * ld + half * 8;
    union { v16bf v; uint4 q[2]; } o;
    o.q[0] = *(const uint4*)(r);        // K = base .. base+7
    o.q[1] = *(const uint4*)(r + 16);   // K = base+16 .. base+23
    return o.v;
}

// Fallback: from row-major f32 source, convert on the fly.
__device__ __forceinline__ v16bf load_afrag_f32(const float* base, int ld, int lane) {
    const int m = lane & 15, half = lane >> 4;
    const float* r = base + (long)m * ld + half * 8;
    float4 a0 = *(const float4*)(r + 0);
    float4 a1 = *(const float4*)(r + 4);
    float4 b0 = *(const float4*)(r + 16);
    float4 b1 = *(const float4*)(r + 20);
    union { v16bf v; unsigned u[8]; } o;
    o.u[0] = pack2(a0.x, a0.y); o.u[1] = pack2(a0.z, a0.w);
    o.u[2] = pack2(a1.x, a1.y); o.u[3] = pack2(a1.z, a1.w);
    o.u[4] = pack2(b0.x, b0.y); o.u[5] = pack2(b0.z, b0.w);
    o.u[6] = pack2(b1.x, b1.y); o.u[7] = pack2(b1.z, b1.w);
    return o.v;
}

// B-fragment (32x16 bf16) from pre-swizzled packed weights:
// tile = nt*KT + kt, per tile 32 lanes x 8 dwords contiguous.
__device__ __forceinline__ v16bf load_bfrag(const unsigned* pk, int tile, int lane) {
    const uint4* p = (const uint4*)(pk + ((long)tile * 256 + lane * 8));
    union { v16bf v; uint4 q[2]; } o;
    o.q[0] = p[0];
    o.q[1] = p[1];
    return o.v;
}

// ---------------------------------------------------------------------------
// Prep: pack W (KxN row-major f32) into per-lane B fragments (bf16).
// ---------------------------------------------------------------------------
__device__ __forceinline__ void pack_one(const float* W, int K, int N,
                                         unsigned* dst, int local) {
    const int tile   = local >> 8;
    const int within = local & 255;
    const int lane   = within >> 3;
    const int j      = within & 7;
    const int KT = K >> 5;
    const int nt = tile / KT;
    const int kt = tile - nt * KT;
    const int k  = kt * 32 + (lane >> 4) * 8 + 2 * j + ((j >> 2) << 3);
    const int n  = nt * 16 + (lane & 15);
    dst[local] = pack2(W[(long)k * N + n], W[(long)(k + 1) * N + n]);
}

__global__ __launch_bounds__(256)
void rnn_prep_pack(const float* __restrict__ Wih, const float* __restrict__ Whh,
                   const float* __restrict__ Wdec, unsigned* __restrict__ ws) {
    const int g = blockIdx.x * 256 + threadIdx.x;
    const int NIH  = (DIN * DLAT) / 2;
    const int NHH  = (DLAT * DLAT) / 2;
    const int NDEC = (DLAT * DOUT) / 2;
    if (g < NIH) {
        pack_one(Wih, DIN, DLAT, ws, g);
    } else if (g < NIH + NHH) {
        pack_one(Whh, DLAT, DLAT, ws + NIH, g - NIH);
    } else if (g < NIH + NHH + NDEC) {
        pack_one(Wdec, DLAT, DOUT, ws + NIH + NHH, g - (NIH + NHH));
    }
}

// One-shot f32 -> row-major bf16 conversion (8 floats / thread, b128 in+out).
__global__ __launch_bounds__(256)
void rnn_cvt_bf16(const float* __restrict__ src, unsigned* __restrict__ dst, long n8) {
    const long i = (long)blockIdx.x * 256 + threadIdx.x;
    if (i >= n8) return;
    const float4* p = (const float4*)(src + i * 8);
    float4 a = p[0], b = p[1];
    uint4 o;
    o.x = pack2(a.x, a.y); o.y = pack2(a.z, a.w);
    o.z = pack2(b.x, b.y); o.w = pack2(b.z, b.w);
    ((uint4*)dst)[i] = o;
}

// ---------------------------------------------------------------------------
// Kernel 1: Z = X @ W_ih (M=65536, K=128, N=512) -> hidden region.
// Block = 256 (8 waves); wave w owns N-tiles 4w..4w+3. Double-buffered K loop.
// ---------------------------------------------------------------------------
template <bool FAST>
__global__ __launch_bounds__(256)
void rnn_encode_gemm(const float* __restrict__ X, const unsigned short* __restrict__ Xbf,
                     const unsigned* __restrict__ pkW, float* __restrict__ Z) {
    const int mt   = blockIdx.x;
    const int lane = threadIdx.x & 31;
    const int w    = threadIdx.x >> 5;
    const int KT   = DIN / 32;
    const long arow = (long)mt * 16 * DIN;
    v8f acc[4] = {};

    v16bf a_cur = FAST ? load_afrag_bf16(Xbf + arow, DIN, lane)
                       : load_afrag_f32(X + arow, DIN, lane);
    v16bf b_cur[4];
    #pragma unroll
    for (int j = 0; j < 4; ++j)
        b_cur[j] = load_bfrag(pkW, (w * 4 + j) * KT, lane);

    #pragma unroll
    for (int kt = 0; kt < KT; ++kt) {
        const int ktn = (kt + 1 < KT) ? kt + 1 : kt;   // harmless re-load on last
        v16bf a_nxt = FAST ? load_afrag_bf16(Xbf + arow + ktn * 32, DIN, lane)
                           : load_afrag_f32(X + arow + ktn * 32, DIN, lane);
        v16bf b_nxt[4];
        #pragma unroll
        for (int j = 0; j < 4; ++j)
            b_nxt[j] = load_bfrag(pkW, (w * 4 + j) * KT + ktn, lane);
        #pragma unroll
        for (int j = 0; j < 4; ++j)
            acc[j] = wmma_bf16(a_cur, b_cur[j], acc[j]);
        a_cur = a_nxt;
        #pragma unroll
        for (int j = 0; j < 4; ++j) b_cur[j] = b_nxt[j];
    }

    const int nl = lane & 15, half = lane >> 4;
    #pragma unroll
    for (int j = 0; j < 4; ++j) {
        const int n = (w * 4 + j) * 16 + nl;
        #pragma unroll
        for (int v = 0; v < 8; ++v) {
            const int m = v + 8 * half;
            Z[(long)(mt * 16 + m) * DLAT + n] = acc[j][v];
        }
    }
}

// ---------------------------------------------------------------------------
// Kernel 2: recurrence. 4 blocks x 512 threads (16 waves); wave w owns
// N-tiles 2w, 2w+1. H in LDS (bf16, row stride 520). Z_t prefetched into
// registers before the GEMM; A/B fragments double-buffered.
// ---------------------------------------------------------------------------
#define LDH 520

template <bool WRITEBF>
__global__ __launch_bounds__(512)
void rnn_recurrence(const float* __restrict__ h0, const unsigned* __restrict__ pkWhh,
                    float* __restrict__ Hid, unsigned short* __restrict__ Hbf,
                    float* __restrict__ hlast) {
    __shared__ __align__(16) unsigned short Hl[16 * LDH];
    const int b0   = blockIdx.x * 16;
    const int tid  = threadIdx.x;
    const int lane = tid & 31;
    const int w    = tid >> 5;
    const int nl   = lane & 15, half = lane >> 4;
    const int KT   = DLAT / 32;

    for (int e = tid; e < 16 * DLAT; e += 512) {
        const int m = e >> 9, n = e & (DLAT - 1);
        Hl[m * LDH + n] = f2bf(h0[(long)(b0 + m) * DLAT + n]);
    }
    __syncthreads();

    long gbase[2][8];   // offset into Hid/Hbf minus the t*DLAT term
    int  lbase[2][8];   // LDS store offsets
    #pragma unroll
    for (int j = 0; j < 2; ++j) {
        const int n = (2 * w + j) * 16 + nl;
        #pragma unroll
        for (int v = 0; v < 8; ++v) {
            const int m = v + 8 * half;
            gbase[j][v] = (long)(b0 + m) * RT * DLAT + n;
            lbase[j][v] = m * LDH + n;
        }
    }

    for (int t = 0; t < RT; ++t) {
        const long toff = (long)t * DLAT;

        // prefetch Z_t — loads issued before the GEMM, consumed after it
        float zreg[2][8];
        #pragma unroll
        for (int j = 0; j < 2; ++j)
            #pragma unroll
            for (int v = 0; v < 8; ++v)
                zreg[j][v] = Hid[gbase[j][v] + toff];

        v8f acc[2] = {};
        v16bf a_cur = load_afrag_bf16(Hl, LDH, lane);
        v16bf b_cur[2];
        #pragma unroll
        for (int j = 0; j < 2; ++j)
            b_cur[j] = load_bfrag(pkWhh, (2 * w + j) * KT, lane);

        #pragma unroll
        for (int kt = 0; kt < KT; ++kt) {
            const int ktn = (kt + 1 < KT) ? kt + 1 : kt;
            v16bf a_nxt = load_afrag_bf16(Hl + ktn * 32, LDH, lane);
            v16bf b_nxt[2];
            #pragma unroll
            for (int j = 0; j < 2; ++j)
                b_nxt[j] = load_bfrag(pkWhh, (2 * w + j) * KT + ktn, lane);
            acc[0] = wmma_bf16(a_cur, b_cur[0], acc[0]);
            acc[1] = wmma_bf16(a_cur, b_cur[1], acc[1]);
            a_cur = a_nxt;
            b_cur[0] = b_nxt[0];
            b_cur[1] = b_nxt[1];
        }

        unsigned short us[2][8];
        #pragma unroll
        for (int j = 0; j < 2; ++j) {
            #pragma unroll
            for (int v = 0; v < 8; ++v) {
                const float pre = acc[j][v] + zreg[j][v];
                const float hv  = 1.0f / (1.0f + __expf(-pre));
                Hid[gbase[j][v] + toff] = hv;                 // hidden_all f32
                us[j][v] = f2bf(hv);
                if (WRITEBF) Hbf[gbase[j][v] + toff] = us[j][v];
            }
        }
        if (t == RT - 1) {   // h_last = h_{T-1}
            #pragma unroll
            for (int j = 0; j < 2; ++j) {
                const int n = (2 * w + j) * 16 + nl;
                #pragma unroll
                for (int v = 0; v < 8; ++v) {
                    const int m = v + 8 * half;
                    hlast[(long)(b0 + m) * DLAT + n] = Hid[gbase[j][v] + toff];
                }
            }
        }
        __syncthreads();   // all waves done reading H_t from LDS
        #pragma unroll
        for (int j = 0; j < 2; ++j)
            #pragma unroll
            for (int v = 0; v < 8; ++v)
                Hl[lbase[j][v]] = us[j][v];
        __syncthreads();   // H_{t+1} visible to all waves
    }
}

// ---------------------------------------------------------------------------
// Kernel 3: output = hidden_all @ W_dec (M=65536, K=512, N=128).
// Block = 128 (4 waves); wave w owns N-tiles 2w, 2w+1. Double-buffered K loop.
// ---------------------------------------------------------------------------
template <bool FAST>
__global__ __launch_bounds__(128)
void rnn_decode_gemm(const float* __restrict__ Hid, const unsigned short* __restrict__ Hbf,
                     const unsigned* __restrict__ pkW, float* __restrict__ Out) {
    const int mt   = blockIdx.x;
    const int lane = threadIdx.x & 31;
    const int w    = threadIdx.x >> 5;
    const int KT   = DLAT / 32;
    const long arow = (long)mt * 16 * DLAT;
    v8f acc[2] = {};

    v16bf a_cur = FAST ? load_afrag_bf16(Hbf + arow, DLAT, lane)
                       : load_afrag_f32(Hid + arow, DLAT, lane);
    v16bf b_cur[2];
    #pragma unroll
    for (int j = 0; j < 2; ++j)
        b_cur[j] = load_bfrag(pkW, (2 * w + j) * KT, lane);

    #pragma unroll
    for (int kt = 0; kt < KT; ++kt) {
        const int ktn = (kt + 1 < KT) ? kt + 1 : kt;
        v16bf a_nxt = FAST ? load_afrag_bf16(Hbf + arow + ktn * 32, DLAT, lane)
                           : load_afrag_f32(Hid + arow + ktn * 32, DLAT, lane);
        v16bf b_nxt[2];
        #pragma unroll
        for (int j = 0; j < 2; ++j)
            b_nxt[j] = load_bfrag(pkW, (2 * w + j) * KT + ktn, lane);
        acc[0] = wmma_bf16(a_cur, b_cur[0], acc[0]);
        acc[1] = wmma_bf16(a_cur, b_cur[1], acc[1]);
        a_cur = a_nxt;
        b_cur[0] = b_nxt[0];
        b_cur[1] = b_nxt[1];
    }

    const int nl = lane & 15, half = lane >> 4;
    #pragma unroll
    for (int j = 0; j < 2; ++j) {
        const int n = (2 * w + j) * 16 + nl;
        #pragma unroll
        for (int v = 0; v < 8; ++v) {
            const int m = v + 8 * half;
            Out[(long)(mt * 16 + m) * DOUT + n] = acc[j][v];
        }
    }
}

// ---------------------------------------------------------------------------
// Workspace layout (dwords)
// ---------------------------------------------------------------------------
static const long PK_IH   = 0;
static const long PK_HH   = PK_IH + (DIN * DLAT) / 2;            //  32768
static const long PK_DEC  = PK_HH + (DLAT * DLAT) / 2;           // +131072
static const long XBF_OFF = PK_DEC + (DLAT * DOUT) / 2;          // +32768
static const long HBF_OFF = XBF_OFF + (long)RB * RT * DIN / 2;   // +4194304
static const long WS_FAST = HBF_OFF + (long)RB * RT * DLAT / 2;  // +16777216 dwords

extern "C" void kernel_launch(void* const* d_in, const int* in_sizes, int n_in,
                              void* d_out, int out_size, void* d_ws, size_t ws_size,
                              hipStream_t stream) {
    const float* x    = (const float*)d_in[0];   // [B,T,DIN]
    const float* h0   = (const float*)d_in[1];   // [B,DLAT]
    const float* Wih  = (const float*)d_in[2];   // [DIN,DLAT]
    const float* Whh  = (const float*)d_in[3];   // [DLAT,DLAT]
    const float* Wdec = (const float*)d_in[4];   // [DLAT,DOUT]

    float* outp  = (float*)d_out;                          // [B,T,DOUT]
    float* hid   = outp + (long)RB * RT * DOUT;            // [B,T,DLAT]
    float* hlast = hid + (long)RB * RT * DLAT;             // [B,DLAT]

    unsigned* ws = (unsigned*)d_ws;
    unsigned*       pkWih  = ws + PK_IH;
    unsigned*       pkWhh  = ws + PK_HH;
    unsigned*       pkWdec = ws + PK_DEC;
    unsigned short* Xbf    = (unsigned short*)(ws + XBF_OFF);
    unsigned short* Hbf    = (unsigned short*)(ws + HBF_OFF);

    const bool fast = ws_size >= (size_t)WS_FAST * 4;

    const int totalDwords = (DIN * DLAT + DLAT * DLAT + DLAT * DOUT) / 2; // 196608
    rnn_prep_pack<<<(totalDwords + 255) / 256, 256, 0, stream>>>(Wih, Whh, Wdec, ws);

    if (fast) {
        const long n8 = (long)RB * RT * DIN / 8;
        rnn_cvt_bf16<<<(int)((n8 + 255) / 256), 256, 0, stream>>>(x, ws + XBF_OFF, n8);
        rnn_encode_gemm<true><<<(RB * RT) / 16, 256, 0, stream>>>(x, Xbf, pkWih, hid);
        rnn_recurrence<true><<<RB / 16, 512, 0, stream>>>(h0, pkWhh, hid, Hbf, hlast);
        rnn_decode_gemm<true><<<(RB * RT) / 16, 128, 0, stream>>>(hid, Hbf, pkWdec, outp);
    } else {
        rnn_encode_gemm<false><<<(RB * RT) / 16, 256, 0, stream>>>(x, Xbf, pkWih, hid);
        rnn_recurrence<false><<<RB / 16, 512, 0, stream>>>(h0, pkWhh, hid, Hbf, hlast);
        rnn_decode_gemm<false><<<(RB * RT) / 16, 128, 0, stream>>>(hid, Hbf, pkWdec, outp);
    }
}
